// _SerialBlock_7919919694496
// MI455X (gfx1250) — matrix operand order repaired and checked by reading, not compiled, not run on hardware
//
#include <hip/hip_runtime.h>
#include <hip/hip_bf16.h>

// ---------------- problem constants (match reference) ----------------
constexpr int BB    = 2;
constexpr int LL    = 2048;
constexpr int DD    = 512;
constexpr int NHA   = 32;
constexpr int HDA   = 16;
constexpr int D_IN  = 1024;
constexpr int NH_M  = 16;
constexpr int PP    = 64;
constexpr int NN    = 64;
constexpr int CONV_DIM = D_IN + 2 * NN;            // 1152
constexpr int D_PROJ   = 2 * D_IN + 2 * NN + NH_M; // 2192
constexpr int KCONV = 4;
constexpr float EPSF = 1e-05f;
constexpr int ROWS = BB * LL;                      // 4096

// ---------------- CDNA5 async global->LDS support (probe via __has_builtin) ----------
#if defined(__has_builtin)
#if __has_builtin(__builtin_amdgcn_global_load_async_to_lds_b128) && \
    __has_builtin(__builtin_amdgcn_global_load_async_to_lds_b32)  && \
    __has_builtin(__builtin_amdgcn_s_wait_asynccnt)
#define HAVE_ASYNC_LDS 1
#endif
#endif
#ifndef HAVE_ASYNC_LDS
#define HAVE_ASYNC_LDS 0
#endif

#if HAVE_ASYNC_LDS
typedef int asy_i32x4 __attribute__((vector_size(16)));  // GCC-style vec, matches builtin proto
#define AS_G __attribute__((address_space(1)))
#define AS_L __attribute__((address_space(3)))
__device__ inline void async_copy16(const void* g, void* l) {
  __builtin_amdgcn_global_load_async_to_lds_b128((AS_G asy_i32x4*)g, (AS_L asy_i32x4*)l, 0, 0);
}
__device__ inline void async_copy4(const void* g, void* l) {
  __builtin_amdgcn_global_load_async_to_lds_b32((AS_G int*)g, (AS_L int*)l, 0, 0);
}
__device__ inline void async_wait0() { __builtin_amdgcn_s_wait_asynccnt(0); }
#else
__device__ inline void async_wait0() {}
#endif

// ---------------- vector types for WMMA ----------------
typedef __attribute__((ext_vector_type(16))) __bf16 bf16x16;
typedef __attribute__((ext_vector_type(8)))  __bf16 bf16x8;
typedef __attribute__((ext_vector_type(8)))  float  f32x8;

union ABf { bf16x16 v; bf16x8 h[2]; };

__device__ inline f32x8 zero_f8() {
  f32x8 z;
#pragma unroll
  for (int i = 0; i < 8; ++i) z[i] = 0.f;
  return z;
}
__device__ inline bf16x8 zero_b8() {
  bf16x8 z;
#pragma unroll
  for (int i = 0; i < 8; ++i) z[i] = (__bf16)0.f;
  return z;
}

// ---------------- fp32 -> bf16 convert ----------------
__global__ void k_f32_to_bf16(const float* __restrict__ x, __bf16* __restrict__ y, long n) {
  long i  = (long)blockIdx.x * blockDim.x + threadIdx.x;
  long st = (long)gridDim.x * blockDim.x;
  for (; i < n; i += st) y[i] = (__bf16)x[i];
}

// flip along sequence dim
__global__ void k_flip_bf16(const __bf16* __restrict__ x, __bf16* __restrict__ y, int Lr, int Dr) {
  long n = (long)BB * Lr * Dr;
  long i = (long)blockIdx.x * blockDim.x + threadIdx.x;
  if (i >= n) return;
  int d = (int)(i % Dr);
  long bl = i / Dr;
  int b = (int)(bl / Lr), l = (int)(bl % Lr);
  y[i] = x[((long)b * Lr + (Lr - 1 - l)) * Dr + d];
}
__global__ void k_flip_f32(const float* __restrict__ x, float* __restrict__ y, int Lr, int Dr) {
  long n = (long)BB * Lr * Dr;
  long i = (long)blockIdx.x * blockDim.x + threadIdx.x;
  if (i >= n) return;
  int d = (int)(i % Dr);
  long bl = i / Dr;
  int b = (int)(bl / Lr), l = (int)(bl % Lr);
  y[i] = x[((long)b * Lr + (Lr - 1 - l)) * Dr + d];
}

// ---------------- WMMA GEMM: C[M,N] = A[M,K]*W[N,K]^T (+bias) ----------------
// block = 256 threads (8 waves); each wave -> 16(M) x 16*NT(N) tile.
// All 8 waves share the same W tiles -> stage them in LDS (double-buffered,
// async global->LDS when available) to cut W global traffic 8x.
template <int NT>
__device__ inline void stage_w(const __bf16* __restrict__ W, __bf16* dst,
                               int n0, int K, int k0, int tid) {
  if (tid < NT * 64) {                 // 16B per thread; row = 64B of W
    int row = tid >> 2, part = tid & 3;
    const __bf16* src = W + (size_t)(n0 + row) * K + k0 + part * 8;
#if HAVE_ASYNC_LDS
    async_copy16(src, dst + row * 32 + part * 8);
#else
    *(bf16x8*)(dst + row * 32 + part * 8) = *(const bf16x8*)src;
#endif
  }
}

template <int NT>
__global__ void k_gemm(const __bf16* __restrict__ A, const __bf16* __restrict__ W,
                       const float* __restrict__ bias, float* __restrict__ C,
                       int M, int N, int K) {
  __shared__ __bf16 wbuf[2][NT * 16 * 32];
  int tid = threadIdx.x;
  int wave = tid >> 5, lane = tid & 31;
  int l = lane & 15, hi = lane >> 4;
  int m0 = blockIdx.y * 128 + wave * 16;
  int n0 = blockIdx.x * (16 * NT);

  f32x8 acc[NT];
#pragma unroll
  for (int j = 0; j < NT; ++j) acc[j] = zero_f8();

  stage_w<NT>(W, &wbuf[0][0], n0, K, 0, tid);

  const __bf16* ap = A + (size_t)(m0 + l) * K + hi * 8;
  for (int k0 = 0; k0 < K; k0 += 32) {
    int buf = (k0 >> 5) & 1;
    async_wait0();
    __syncthreads();
    if (k0 + 32 < K) stage_w<NT>(W, &wbuf[buf ^ 1][0], n0, K, k0 + 32, tid);

    ABf a;
    a.h[0] = *(const bf16x8*)(ap + k0);
    a.h[1] = *(const bf16x8*)(ap + k0 + 16);
    __builtin_prefetch(ap + k0 + 128, 0, 0);
#pragma unroll
    for (int j = 0; j < NT; ++j) {
      const __bf16* bp = &wbuf[buf][(j * 16 + l) * 32 + hi * 8];
      ABf b;
      b.h[0] = *(const bf16x8*)bp;
      b.h[1] = *(const bf16x8*)(bp + 16);
      acc[j] = __builtin_amdgcn_wmma_f32_16x16x32_bf16(false, a.v, false, b.v,
                                                       (short)0, acc[j], false, false);
    }
  }
#pragma unroll
  for (int j = 0; j < NT; ++j) {
    int n = n0 + j * 16 + l;
    float bv = bias ? bias[n] : 0.f;
#pragma unroll
    for (int r = 0; r < 8; ++r) {
      C[(size_t)(m0 + hi * 8 + r) * N + n] = acc[j][r] + bv;
    }
  }
}

// ---------------- qkv split: fp32 [B,L,3D] -> q/k [B,NH,L,16] bf16, vT [B,NH,16,L]
__global__ void k_qkv_prep(const float* __restrict__ qkv, __bf16* __restrict__ q,
                           __bf16* __restrict__ k, __bf16* __restrict__ vT) {
  long i = (long)blockIdx.x * blockDim.x + threadIdx.x;
  if (i >= (long)ROWS * DD) return;
  int d = (int)(i & 15);
  int h = (int)((i >> 4) & 31);
  long bl = i >> 9;
  int b = (int)(bl / LL), lpos = (int)(bl % LL);
  const float* base = qkv + bl * (3 * DD);
  float qv = base[h * 16 + d];
  float kv = base[DD + h * 16 + d];
  float vv = base[2 * DD + h * 16 + d];
  size_t o = (((size_t)b * NHA + h) * LL + lpos) * HDA + d;
  q[o] = (__bf16)(qv * 0.25f);  // 1/sqrt(16) folded into Q
  k[o] = (__bf16)kv;
  vT[(((size_t)b * NHA + h) * HDA + d) * LL + lpos] = (__bf16)vv;
}

// ---------------- flash attention (non-causal), 64 keys per outer step ----------------
// grid = (L/128, B*NHA), block = 256 (8 waves); each wave owns 16 queries.
__global__ void k_attn(const __bf16* __restrict__ q, const __bf16* __restrict__ kk,
                       const __bf16* __restrict__ vT, float* __restrict__ o) {
  __shared__ __bf16 pbuf[8][4 * 256];  // per-wave 4x 16x16 P tiles
  int wave = threadIdx.x >> 5, lane = threadIdx.x & 31;
  int l = lane & 15, hi = lane >> 4;
  int bh = blockIdx.y;
  int b = bh >> 5, h = bh & 31;
  size_t base = (size_t)bh * LL * HDA;
  int q0 = blockIdx.x * 128 + wave * 16;

  ABf qa;
  qa.h[0] = *(const bf16x8*)(q + base + (size_t)(q0 + l) * HDA + hi * 8);
  qa.h[1] = zero_b8();  // K-dim 16..31 zero-padded

  float mrow[8], lrow[8];
  f32x8 oacc = zero_f8();
#pragma unroll
  for (int r = 0; r < 8; ++r) { mrow[r] = -3.0e38f; lrow[r] = 0.f; }

  __bf16* ps = &pbuf[wave][0];

  for (int jo = 0; jo < LL / 64; ++jo) {
    // ---- 4x QK^T WMMA (back-to-back) ----
    f32x8 sl[4];
#pragma unroll
    for (int ji = 0; ji < 4; ++ji) {
      int j = jo * 4 + ji;
      ABf kb;
      {
        const __bf16* kp = kk + base + (size_t)(j * 16 + l) * HDA;
        kb.h[0] = *(const bf16x8*)(kp + hi * 8);
        kb.h[1] = zero_b8();
      }
      sl[ji] = __builtin_amdgcn_wmma_f32_16x16x32_bf16(false, qa.v, false, kb.v,
                                                       (short)0, zero_f8(), false, false);
    }
    // ---- one online-softmax update per 64 keys ----
#pragma unroll
    for (int r = 0; r < 8; ++r) {
      float mx = fmaxf(fmaxf(sl[0][r], sl[1][r]), fmaxf(sl[2][r], sl[3][r]));
      mx = fmaxf(mx, __shfl_xor(mx, 1, 32));
      mx = fmaxf(mx, __shfl_xor(mx, 2, 32));
      mx = fmaxf(mx, __shfl_xor(mx, 4, 32));
      mx = fmaxf(mx, __shfl_xor(mx, 8, 32));
      float mn = fmaxf(mrow[r], mx);
      float al = __expf(mrow[r] - mn);
      mrow[r] = mn;
      float rs = 0.f;
#pragma unroll
      for (int ji = 0; ji < 4; ++ji) {
        float p = __expf(sl[ji][r] - mn);
        sl[ji][r] = p;
        rs += p;
      }
      rs += __shfl_xor(rs, 1, 32);
      rs += __shfl_xor(rs, 2, 32);
      rs += __shfl_xor(rs, 4, 32);
      rs += __shfl_xor(rs, 8, 32);
      lrow[r] = lrow[r] * al + rs;
      oacc[r] *= al;
#pragma unroll
      for (int ji = 0; ji < 4; ++ji)
        ps[ji * 256 + (hi * 8 + r) * 16 + l] = (__bf16)sl[ji][r];
    }
    // ---- 4x PV WMMA (P transposed through per-wave LDS slot) ----
#pragma unroll
    for (int ji = 0; ji < 4; ++ji) {
      int j = jo * 4 + ji;
      ABf pa;
      pa.h[0] = *(const bf16x8*)(ps + ji * 256 + l * 16 + hi * 8);
      pa.h[1] = zero_b8();
      ABf vb;
      {
        const __bf16* vp = vT + (size_t)bh * HDA * LL + (size_t)l * LL + j * 16;
        vb.h[0] = *(const bf16x8*)(vp + hi * 8);
        vb.h[1] = zero_b8();
      }
      oacc = __builtin_amdgcn_wmma_f32_16x16x32_bf16(false, pa.v, false, vb.v,
                                                     (short)0, oacc, false, false);
    }
  }
#pragma unroll
  for (int r = 0; r < 8; ++r) {
    int m = q0 + hi * 8 + r;
    o[((size_t)b * LL + m) * DD + h * 16 + l] = oacc[r] / lrow[r];
  }
}

// ---------------- add(+add) + LayerNorm over D=512, optional bf16 copy ----------------
__global__ void k_add_ln(const float* __restrict__ x0, const float* __restrict__ x1,
                         const float* __restrict__ x2, const float* __restrict__ g,
                         const float* __restrict__ bb, float* __restrict__ outf,
                         __bf16* __restrict__ outb) {
  int row = blockIdx.x, t = threadIdx.x;
  __shared__ float red[8];
  size_t off = (size_t)row * DD;
  float v[2];
#pragma unroll
  for (int e = 0; e < 2; ++e) {
    int c = t + e * 256;
    float s = x0[off + c] + x1[off + c];
    if (x2) s += x2[off + c];
    v[e] = s;
  }
  float sum = v[0] + v[1];
  for (int d2 = 1; d2 < 32; d2 <<= 1) sum += __shfl_xor(sum, d2, 32);
  if ((t & 31) == 0) red[t >> 5] = sum;
  __syncthreads();
  float tot = 0.f;
#pragma unroll
  for (int i = 0; i < 8; ++i) tot += red[i];
  float mean = tot / (float)DD;
  __syncthreads();
  float sq = (v[0] - mean) * (v[0] - mean) + (v[1] - mean) * (v[1] - mean);
  for (int d2 = 1; d2 < 32; d2 <<= 1) sq += __shfl_xor(sq, d2, 32);
  if ((t & 31) == 0) red[t >> 5] = sq;
  __syncthreads();
  float vt = 0.f;
#pragma unroll
  for (int i = 0; i < 8; ++i) vt += red[i];
  float inv = rsqrtf(vt / (float)DD + EPSF);
#pragma unroll
  for (int e = 0; e < 2; ++e) {
    int c = t + e * 256;
    float r = (v[e] - mean) * inv * g[c] + bb[c];
    outf[off + c] = r;
    if (outb) outb[off + c] = (__bf16)r;
  }
}

// ---------------- mamba: dt = softplus(zx[...,-16:]+bias), dA = exp(dt * -exp(A_log))
__global__ void k_dtdA(const float* __restrict__ zx, const float* __restrict__ dt_bias,
                       const float* __restrict__ A_log, float* __restrict__ dt,
                       float* __restrict__ dA) {
  int i = blockIdx.x * blockDim.x + threadIdx.x;
  if (i >= ROWS * NH_M) return;
  int h = i & 15;
  long row = i >> 4;
  float x = zx[row * D_PROJ + (D_PROJ - NH_M) + h] + dt_bias[h];
  float dtv = (x > 20.f) ? x : log1pf(__expf(x));
  float A = -__expf(A_log[h]);
  dt[i] = dtv;
  dA[i] = __expf(dtv * A);
}

// ---------------- depthwise causal conv (K=4) + SiLU ----------------
__global__ void k_conv(const float* __restrict__ zx, const float* __restrict__ convw,
                       const float* __restrict__ convb, float* __restrict__ xBC) {
  long i = (long)blockIdx.x * blockDim.x + threadIdx.x;
  if (i >= (long)ROWS * CONV_DIM) return;
  int c = (int)(i % CONV_DIM);
  long bl = i / CONV_DIM;
  int b = (int)(bl / LL), l = (int)(bl % LL);
  float acc = convb[c];
#pragma unroll
  for (int j = 0; j < KCONV; ++j) {
    int pos = l - (KCONV - 1) + j;
    if (pos >= 0)
      acc += zx[(((long)b * LL) + pos) * D_PROJ + D_IN + c] * convw[c * KCONV + j];
  }
  xBC[i] = acc / (1.f + __expf(-acc));  // SiLU
}

// ---------------- sequential SSM scan with async double-buffered B/C ----------------
// grid = B*NH_M, block = 64 (thread = p-row; state row of 64 kept in VGPRs)
__global__ void k_scan(const float* __restrict__ xBC, const float* __restrict__ dt,
                       const float* __restrict__ dA, float* __restrict__ ys) {
  int bh = blockIdx.x;
  int b = bh >> 4, h = bh & 15;
  int p = threadIdx.x;  // 0..63
  float s[NN];
#pragma unroll
  for (int n = 0; n < NN; ++n) s[n] = 0.f;
  __shared__ float Bbuf[2][NN], Cbuf[2][NN];

  // prefetch step 0
  {
    const float* xrow0 = xBC + ((long)b * LL) * CONV_DIM;
#if HAVE_ASYNC_LDS
    async_copy4(xrow0 + D_IN + p, &Bbuf[0][p]);
    async_copy4(xrow0 + D_IN + NN + p, &Cbuf[0][p]);
#else
    Bbuf[0][p] = xrow0[D_IN + p];
    Cbuf[0][p] = xrow0[D_IN + NN + p];
#endif
  }
  for (int l = 0; l < LL; ++l) {
    int buf = l & 1;
    long row = (long)b * LL + l;
    async_wait0();
    __syncthreads();
    if (l + 1 < LL) {  // prefetch step l+1 while computing step l
      const float* xrn = xBC + (row + 1) * CONV_DIM;
#if HAVE_ASYNC_LDS
      async_copy4(xrn + D_IN + p, &Bbuf[buf ^ 1][p]);
      async_copy4(xrn + D_IN + NN + p, &Cbuf[buf ^ 1][p]);
#else
      Bbuf[buf ^ 1][p] = xrn[D_IN + p];
      Cbuf[buf ^ 1][p] = xrn[D_IN + NN + p];
#endif
    }
    float dtv = dt[row * NH_M + h];
    float dav = dA[row * NH_M + h];
    float dtx = dtv * xBC[row * CONV_DIM + h * PP + p];
    float acc = 0.f;
#pragma unroll
    for (int n = 0; n < NN; ++n) {
      s[n] = s[n] * dav + dtx * Bbuf[buf][n];
      acc += s[n] * Cbuf[buf][n];
    }
    ys[(row * NH_M + h) * PP + p] = acc;
  }
}

// ---------------- y = (ys + D*x) * silu(z); RMS-norm(1024)*norm_w -> bf16 -------------
__global__ void k_gatenorm(const float* __restrict__ ys, const float* __restrict__ xBC,
                           const float* __restrict__ zx, const float* __restrict__ Dp,
                           const float* __restrict__ norm_w, __bf16* __restrict__ yout) {
  int row = blockIdx.x, t = threadIdx.x;
  __shared__ float red[8];
  float v[4];
#pragma unroll
  for (int e = 0; e < 4; ++e) {
    int c = t + e * 256;
    int h = c >> 6;
    float x = xBC[(size_t)row * CONV_DIM + c];
    float y = ys[(size_t)row * D_IN + c] + Dp[h] * x;
    float z = zx[(size_t)row * D_PROJ + c];
    y *= z / (1.f + __expf(-z));
    v[e] = y;
  }
  float ss = v[0] * v[0] + v[1] * v[1] + v[2] * v[2] + v[3] * v[3];
  for (int d2 = 1; d2 < 32; d2 <<= 1) ss += __shfl_xor(ss, d2, 32);
  if ((t & 31) == 0) red[t >> 5] = ss;
  __syncthreads();
  float tot = 0.f;
#pragma unroll
  for (int i = 0; i < 8; ++i) tot += red[i];
  float inv = rsqrtf(tot / (float)D_IN + EPSF);
#pragma unroll
  for (int e = 0; e < 4; ++e) {
    int c = t + e * 256;
    yout[(size_t)row * D_IN + c] = (__bf16)(v[e] * inv * norm_w[c]);
  }
}

// ======================= host side =======================
static inline void* wsalloc(char*& p, size_t bytes) {
  void* r = (void*)p;
  p += (bytes + 255) & ~(size_t)255;
  return r;
}

static void launch_cvt(const float* x, __bf16* y, long n, hipStream_t s) {
  int blocks = (int)((n + 255) / 256);
  if (blocks > 32768) blocks = 32768;
  k_f32_to_bf16<<<blocks, 256, 0, s>>>(x, y, n);
}

static void run_mamba(const __bf16* hin_bf, const __bf16* Wi_bf, const float* convw,
                      const float* convb, const float* dt_bias, const float* A_log,
                      const float* Dp, const float* norm_w, const __bf16* Wo_bf,
                      float* zx, float* xBC, float* dtb, float* dAb, float* ys,
                      __bf16* ynorm, float* outp, hipStream_t stream) {
  k_gemm<1><<<dim3(D_PROJ / 16, ROWS / 128), 256, 0, stream>>>(hin_bf, Wi_bf, nullptr, zx,
                                                               ROWS, D_PROJ, DD);
  k_dtdA<<<(ROWS * NH_M + 255) / 256, 256, 0, stream>>>(zx, dt_bias, A_log, dtb, dAb);
  k_conv<<<(int)(((long)ROWS * CONV_DIM + 255) / 256), 256, 0, stream>>>(zx, convw, convb, xBC);
  k_scan<<<BB * NH_M, 64, 0, stream>>>(xBC, dtb, dAb, ys);
  k_gatenorm<<<ROWS, 256, 0, stream>>>(ys, xBC, zx, Dp, norm_w, ynorm);
  k_gemm<4><<<dim3(DD / 64, ROWS / 128), 256, 0, stream>>>(ynorm, Wo_bf, nullptr, outp,
                                                           ROWS, DD, D_IN);
}

extern "C" void kernel_launch(void* const* d_in, const int* in_sizes, int n_in,
                              void* d_out, int out_size, void* d_ws, size_t ws_size,
                              hipStream_t stream) {
  const float* h      = (const float*)d_in[0];
  const float* Wqkv   = (const float*)d_in[1];
  const float* bqkv   = (const float*)d_in[2];
  const float* Wo_a   = (const float*)d_in[3];
  const float* bo_a   = (const float*)d_in[4];
  const float* g1     = (const float*)d_in[5];
  const float* b1     = (const float*)d_in[6];
  const float* g2     = (const float*)d_in[7];
  const float* b2     = (const float*)d_in[8];
  const float* f_Wi     = (const float*)d_in[9];
  const float* f_convw  = (const float*)d_in[10];
  const float* f_convb  = (const float*)d_in[11];
  const float* f_dtb    = (const float*)d_in[12];
  const float* f_Alog   = (const float*)d_in[13];
  const float* f_D      = (const float*)d_in[14];
  const float* f_nw     = (const float*)d_in[15];
  const float* f_Wo     = (const float*)d_in[16];
  const float* b_Wi     = (const float*)d_in[17];
  const float* b_convw  = (const float*)d_in[18];
  const float* b_convb  = (const float*)d_in[19];
  const float* b_dtb    = (const float*)d_in[20];
  const float* b_Alog   = (const float*)d_in[21];
  const float* b_D      = (const float*)d_in[22];
  const float* b_nw     = (const float*)d_in[23];
  const float* b_Wo     = (const float*)d_in[24];
  float* out = (float*)d_out;

  char* wp = (char*)d_ws;
  __bf16* h_bf    = (__bf16*)wsalloc(wp, (size_t)ROWS * DD * 2);
  __bf16* Wqkv_bf = (__bf16*)wsalloc(wp, (size_t)3 * DD * DD * 2);
  __bf16* Woa_bf  = (__bf16*)wsalloc(wp, (size_t)DD * DD * 2);
  __bf16* fWi_bf  = (__bf16*)wsalloc(wp, (size_t)D_PROJ * DD * 2);
  __bf16* fWo_bf  = (__bf16*)wsalloc(wp, (size_t)DD * D_IN * 2);
  __bf16* bWi_bf  = (__bf16*)wsalloc(wp, (size_t)D_PROJ * DD * 2);
  __bf16* bWo_bf  = (__bf16*)wsalloc(wp, (size_t)DD * D_IN * 2);
  float*  qkv     = (float*)wsalloc(wp, (size_t)ROWS * 3 * DD * 4);
  __bf16* qb      = (__bf16*)wsalloc(wp, (size_t)ROWS * DD * 2);
  __bf16* kb      = (__bf16*)wsalloc(wp, (size_t)ROWS * DD * 2);
  __bf16* vTb     = (__bf16*)wsalloc(wp, (size_t)ROWS * DD * 2);
  float*  attn    = (float*)wsalloc(wp, (size_t)ROWS * DD * 4);
  __bf16* attn_bf = (__bf16*)wsalloc(wp, (size_t)ROWS * DD * 2);
  float*  aproj   = (float*)wsalloc(wp, (size_t)ROWS * DD * 4);
  float*  hn      = (float*)wsalloc(wp, (size_t)ROWS * DD * 4);
  __bf16* hn_bf   = (__bf16*)wsalloc(wp, (size_t)ROWS * DD * 2);
  __bf16* hnf_bf  = (__bf16*)wsalloc(wp, (size_t)ROWS * DD * 2);
  float*  zx      = (float*)wsalloc(wp, (size_t)ROWS * D_PROJ * 4);
  float*  xBC     = (float*)wsalloc(wp, (size_t)ROWS * CONV_DIM * 4);
  float*  dtb     = (float*)wsalloc(wp, (size_t)ROWS * NH_M * 4);
  float*  dAb     = (float*)wsalloc(wp, (size_t)ROWS * NH_M * 4);
  float*  ys      = (float*)wsalloc(wp, (size_t)ROWS * D_IN * 4);
  __bf16* ynorm   = (__bf16*)wsalloc(wp, (size_t)ROWS * D_IN * 2);
  float*  hf      = (float*)wsalloc(wp, (size_t)ROWS * DD * 4);
  float*  hbt     = (float*)wsalloc(wp, (size_t)ROWS * DD * 4);
  float*  hb      = (float*)wsalloc(wp, (size_t)ROWS * DD * 4);

  // 1) bf16 weight/activation copies
  launch_cvt(h, h_bf, (long)ROWS * DD, stream);
  launch_cvt(Wqkv, Wqkv_bf, (long)3 * DD * DD, stream);
  launch_cvt(Wo_a, Woa_bf, (long)DD * DD, stream);
  launch_cvt(f_Wi, fWi_bf, (long)D_PROJ * DD, stream);
  launch_cvt(f_Wo, fWo_bf, (long)DD * D_IN, stream);
  launch_cvt(b_Wi, bWi_bf, (long)D_PROJ * DD, stream);
  launch_cvt(b_Wo, bWo_bf, (long)DD * D_IN, stream);

  // 2) qkv = h @ Wqkv^T + bqkv
  k_gemm<4><<<dim3((3 * DD) / 64, ROWS / 128), 256, 0, stream>>>(h_bf, Wqkv_bf, bqkv, qkv,
                                                                 ROWS, 3 * DD, DD);
  // 3) split/scale/transpose per-head
  k_qkv_prep<<<(int)(((long)ROWS * DD + 255) / 256), 256, 0, stream>>>(qkv, qb, kb, vTb);

  // 4) attention
  k_attn<<<dim3(LL / 128, BB * NHA), 256, 0, stream>>>(qb, kb, vTb, attn);

  // 5) out-proj
  launch_cvt(attn, attn_bf, (long)ROWS * DD, stream);
  k_gemm<4><<<dim3(DD / 64, ROWS / 128), 256, 0, stream>>>(attn_bf, Woa_bf, bo_a, aproj,
                                                           ROWS, DD, DD);
  // 6) hn = LN(h + aproj)
  k_add_ln<<<ROWS, 256, 0, stream>>>(h, aproj, nullptr, g1, b1, hn, hn_bf);

  // 7) forward mamba
  run_mamba(hn_bf, fWi_bf, f_convw, f_convb, f_dtb, f_Alog, f_D, f_nw, fWo_bf,
            zx, xBC, dtb, dAb, ys, ynorm, hf, stream);

  // 8) backward mamba on flipped sequence, flip result back
  k_flip_bf16<<<(int)(((long)ROWS * DD + 255) / 256), 256, 0, stream>>>(hn_bf, hnf_bf, LL, DD);
  run_mamba(hnf_bf, bWi_bf, b_convw, b_convb, b_dtb, b_Alog, b_D, b_nw, bWo_bf,
            zx, xBC, dtb, dAb, ys, ynorm, hbt, stream);
  k_flip_f32<<<(int)(((long)ROWS * DD + 255) / 256), 256, 0, stream>>>(hbt, hb, LL, DD);

  // 9) out = LN(hn + hf + hb)
  k_add_ln<<<ROWS, 256, 0, stream>>>(hn, hf, hb, g2, b2, out, nullptr);
}
